// ContactATT_43164421325048
// MI455X (gfx1250) — matrix operand order repaired and checked by reading, not compiled, not run on hardware
//
#include <hip/hip_runtime.h>
#include <hip/hip_bf16.h>

// Problem constants (from reference)
constexpr int BB = 16;
constexpr int LQ = 2048;
constexpr int LK = 2048;
constexpr int DD = 256;

typedef __bf16 v16bf __attribute__((ext_vector_type(16)));
typedef float  v8f   __attribute__((ext_vector_type(8)));

union BF16Frag {
    v16bf v;
    unsigned short u[16];
    uint4 q2[2];
};

__device__ __forceinline__ unsigned short f2bf(float f) {
    unsigned int u = __float_as_uint(f);
    unsigned int r = u + 0x7FFFu + ((u >> 16) & 1u);   // round-to-nearest-even
    return (unsigned short)(r >> 16);
}
__device__ __forceinline__ float bf2f(unsigned short h) {
    return __uint_as_float(((unsigned int)h) << 16);
}
__device__ __forceinline__ void pack8(unsigned short* dst, float4 a, float4 b) {
    dst[0] = f2bf(a.x); dst[1] = f2bf(a.y); dst[2] = f2bf(a.z); dst[3] = f2bf(a.w);
    dst[4] = f2bf(b.x); dst[5] = f2bf(b.y); dst[6] = f2bf(b.z); dst[7] = f2bf(b.w);
}

__device__ __forceinline__ v8f wmma_bf16(const BF16Frag& a, const BF16Frag& b, v8f c) {
    return __builtin_amdgcn_wmma_f32_16x16x32_bf16(
        /*neg_a=*/false, a.v, /*neg_b=*/false, b.v,
        /*c_mod=*/(short)0, c, /*reuse_a=*/false, /*reuse_b=*/false);
}

// ---------------------------------------------------------------------------
// Kernel 1: transpose W [e][d] f32 -> Wt [d][e] bf16 (3 matrices)
// ---------------------------------------------------------------------------
__global__ __launch_bounds__(256)
void wtrans_kernel(const float* __restrict__ Wq, const float* __restrict__ Wk,
                   const float* __restrict__ Wv,
                   unsigned short* __restrict__ Wtq, unsigned short* __restrict__ Wtk,
                   unsigned short* __restrict__ Wtv) {
    int idx = blockIdx.x * 256 + threadIdx.x;          // [0, 3*DD*DD)
    int which = idx / (DD * DD);
    int r = idx % (DD * DD);
    int d = r / DD, e = r % DD;
    const float* W = (which == 0) ? Wq : (which == 1) ? Wk : Wv;
    unsigned short* Wt = (which == 0) ? Wtq : (which == 1) ? Wtk : Wtv;
    Wt[d * DD + e] = f2bf(W[e * DD + d]);
}

// ---------------------------------------------------------------------------
// Kernel 2: projections. out[l][e] = sum_d In[l][d] * W[e][d]  (= In @ Wt)
// One wave computes a 16x16 tile with 8 WMMA steps over D=256.
// which==0 -> Q (row-major bf16); which==1 -> Kt (transposed bf16); which==2 -> V
// ---------------------------------------------------------------------------
__global__ __launch_bounds__(32)
void proj_kernel(const float* __restrict__ x, const float* __restrict__ y,
                 const unsigned short* __restrict__ Wtq,
                 const unsigned short* __restrict__ Wtk,
                 const unsigned short* __restrict__ Wtv,
                 unsigned short* __restrict__ Qbf,
                 unsigned short* __restrict__ Ktbf,
                 unsigned short* __restrict__ Vbf) {
    const int lane = threadIdx.x;
    const int e0 = blockIdx.x * 16;
    const int l0 = blockIdx.y * 16;
    const int which = blockIdx.z % 3;
    const int b = blockIdx.z / 3;

    const float* In = ((which == 0) ? x : y) + (size_t)b * LQ * DD;
    const unsigned short* Wt = (which == 0) ? Wtq : (which == 1) ? Wtk : Wtv;

    const int m   = lane & 15;
    const int hi  = lane >> 4;
    const int off = hi * 8;           // K offset for this lane half (ISA A layout)
    const float* arow = In + (size_t)(l0 + m) * DD;

    v8f acc = {};
    for (int s = 0; s < DD / 32; ++s) {
        const int k0 = s * 32;
        BF16Frag a;
        float4 f0 = *(const float4*)(arow + k0 + off);
        float4 f1 = *(const float4*)(arow + k0 + off + 4);
        float4 f2 = *(const float4*)(arow + k0 + 16 + off);
        float4 f3 = *(const float4*)(arow + k0 + 20 + off);
        pack8(a.u, f0, f1);
        pack8(a.u + 8, f2, f3);

        BF16Frag bf;
        const unsigned short* brow = Wt + (size_t)(k0 + off) * DD + e0 + m;
        for (int j = 0; j < 8; ++j) { bf.u[j] = brow[(size_t)j * DD]; bf.u[8 + j] = brow[(size_t)(16 + j) * DD]; }

        acc = wmma_bf16(a, bf, acc);
    }

    const int n = lane & 15;
    const int mbase = hi * 8;
    unsigned short h[8];
#pragma unroll
    for (int j = 0; j < 8; ++j) h[j] = f2bf(acc[j]);

    if (which == 1) {
        // K stored transposed: Kt[b][e][kk]
        unsigned short* dst = Ktbf + (size_t)b * DD * LK + (size_t)(e0 + n) * LK + l0 + mbase;
#pragma unroll
        for (int j = 0; j < 8; ++j) dst[j] = h[j];
    } else {
        unsigned short* O = ((which == 0) ? Qbf : Vbf) +
                            (size_t)b * LQ * DD + (size_t)(l0 + mbase) * DD + (e0 + n);
#pragma unroll
        for (int j = 0; j < 8; ++j) O[(size_t)j * DD] = h[j];
    }
}

// ---------------------------------------------------------------------------
// Kernel 3: qn[b*LQ+q] = ||q_row||^2 (block of 256 per row, D=256)
// ---------------------------------------------------------------------------
__global__ __launch_bounds__(256)
void qnorm_kernel(const unsigned short* __restrict__ Qbf, float* __restrict__ qn) {
    __shared__ float red[256];
    const size_t row = blockIdx.x;           // b*LQ + q
    const int t = threadIdx.x;
    float v = bf2f(Qbf[row * DD + t]);
    red[t] = v * v;
    __syncthreads();
    for (int s = 128; s > 0; s >>= 1) {
        if (t < s) red[t] += red[t + s];
        __syncthreads();
    }
    if (t == 0) qn[row] = red[0];
}

// ---------------------------------------------------------------------------
// Kernel 4: kn[b*LK+kk] = ||k_row||^2 from Kt (coalesced over kk)
// ---------------------------------------------------------------------------
__global__ __launch_bounds__(256)
void knorm_kernel(const unsigned short* __restrict__ Ktbf, float* __restrict__ kn) {
    const int idx = blockIdx.x * 256 + threadIdx.x;    // b*LK + kk
    const int b = idx / LK, kk = idx % LK;
    const unsigned short* base = Ktbf + (size_t)b * DD * LK + kk;
    float s = 0.0f;
    for (int d = 0; d < DD; ++d) {
        float v = bf2f(base[(size_t)d * LK]);
        s += v * v;
    }
    kn[idx] = s;
}

// ---------------------------------------------------------------------------
// Kernel 5: scores. One wave: 16 q-rows x 128 keys (8 WMMA tiles, A reused).
// p = exp(-sqrt(max(qn+kn-2*(q.k),0))); masked -> 0. Writes UNNORMALIZED attn.
// ---------------------------------------------------------------------------
__global__ __launch_bounds__(32)
void scores_kernel(const unsigned short* __restrict__ Qbf,
                   const unsigned short* __restrict__ Ktbf,
                   const float* __restrict__ qn, const float* __restrict__ kn,
                   const unsigned char* __restrict__ mask,
                   float* __restrict__ attn) {
    const int lane = threadIdx.x;
    const int kk0 = blockIdx.x * 128;
    const int q0 = blockIdx.y * 16;
    const int b = blockIdx.z;
    const int m = lane & 15;
    const int hi = lane >> 4;
    const int off = hi * 8;

    const unsigned short* arow = Qbf + (size_t)b * LQ * DD + (size_t)(q0 + m) * DD;
    const unsigned short* Kbase = Ktbf + (size_t)b * DD * LK + kk0;

    v8f acc[8];
#pragma unroll
    for (int t = 0; t < 8; ++t) acc[t] = (v8f){};

    for (int s = 0; s < DD / 32; ++s) {
        const int k0 = s * 32;
        BF16Frag a;
        a.q2[0] = *(const uint4*)(arow + k0 + off);
        a.q2[1] = *(const uint4*)(arow + k0 + 16 + off);

        const unsigned short* brow = Kbase + (size_t)(k0 + off) * LK + m;
#pragma unroll
        for (int t = 0; t < 8; ++t) {
            BF16Frag bf;
            for (int j = 0; j < 8; ++j) { bf.u[j] = brow[(size_t)j * LK + t * 16]; bf.u[8 + j] = brow[(size_t)(16 + j) * LK + t * 16]; }
            acc[t] = wmma_bf16(a, bf, acc[t]);
        }
    }

    const int n = lane & 15;
    const int mbase = hi * 8;
    float qarr[8];
#pragma unroll
    for (int j = 0; j < 8; ++j) qarr[j] = qn[(size_t)b * LQ + q0 + mbase + j];
    const size_t attnbase = (size_t)b * LQ * LK;

#pragma unroll
    for (int t = 0; t < 8; ++t) {
        const int kc = kk0 + t * 16 + n;
        const float knv = kn[(size_t)b * LK + kc];
        const bool msk = mask[(size_t)b * LK + kc] != 0;
#pragma unroll
        for (int j = 0; j < 8; ++j) {
            float d2 = qarr[j] + knv - 2.0f * acc[t][j];
            float dist = sqrtf(fmaxf(d2, 0.0f));
            float p = __expf(-dist);      // in (0,1]: softmax needs no max pass
            if (msk) p = 0.0f;            // exp(-1e10) == 0 exactly
            attn[attnbase + (size_t)(q0 + mbase + j) * LK + kc] = p;
        }
    }
}

// ---------------------------------------------------------------------------
// Kernel 6: per-row sum then normalize attn in place (softmax denominator)
// ---------------------------------------------------------------------------
__global__ __launch_bounds__(256)
void rownorm_kernel(float* __restrict__ attn) {
    __shared__ float red[256];
    __shared__ float sinv;
    const size_t row = blockIdx.x;           // b*LQ + q
    float* p = attn + row * (size_t)LK;
    const int t = threadIdx.x;
    float s = 0.0f;
    for (int i = t; i < LK; i += 256) s += p[i];
    red[t] = s;
    __syncthreads();
    for (int w = 128; w > 0; w >>= 1) {
        if (t < w) red[t] += red[t + w];
        __syncthreads();
    }
    if (t == 0) sinv = (red[0] > 0.0f) ? 1.0f / red[0] : 0.0f;
    __syncthreads();
    const float inv = sinv;
    for (int i = t; i < LK; i += 256) p[i] *= inv;
}

// ---------------------------------------------------------------------------
// Kernel 7: att_out = attn @ V. One wave: 16 q-rows x FULL D=256 (16 WMMA
// accumulator tiles, 128 acc VGPRs/lane) so attn is streamed from HBM exactly
// once. V (1 MB bf16 per batch) stays L2-resident across the whole q loop.
// ---------------------------------------------------------------------------
__global__ __launch_bounds__(32)
void av_kernel(const float* __restrict__ attn, const unsigned short* __restrict__ Vbf,
               float* __restrict__ out) {
    const int lane = threadIdx.x;
    const int q0 = blockIdx.x * 16;
    const int b = blockIdx.y;
    const int m = lane & 15;
    const int hi = lane >> 4;
    const int off = hi * 8;

    const float* arow = attn + (size_t)b * LQ * LK + (size_t)(q0 + m) * LK;
    const unsigned short* Vb = Vbf + (size_t)b * LK * DD;

    v8f acc[16];
#pragma unroll
    for (int t = 0; t < 16; ++t) acc[t] = (v8f){};

    for (int s = 0; s < LK / 32; ++s) {
        const int k0 = s * 32;
        // Prefetch the next attn chunk of this row (gfx1250 global_prefetch_b8)
        __builtin_prefetch((const void*)(arow + k0 + 512), 0, 0);

        BF16Frag a;
        float4 f0 = *(const float4*)(arow + k0 + off);
        float4 f1 = *(const float4*)(arow + k0 + off + 4);
        float4 f2 = *(const float4*)(arow + k0 + 16 + off);
        float4 f3 = *(const float4*)(arow + k0 + 20 + off);
        pack8(a.u, f0, f1);
        pack8(a.u + 8, f2, f3);

        const unsigned short* brow = Vb + (size_t)(k0 + off) * DD + m;
#pragma unroll
        for (int t = 0; t < 16; ++t) {
            BF16Frag bf;
            for (int j = 0; j < 8; ++j) { bf.u[j] = brow[(size_t)j * DD + t * 16]; bf.u[8 + j] = brow[(size_t)(16 + j) * DD + t * 16]; }
            acc[t] = wmma_bf16(a, bf, acc[t]);
        }
    }

    const int n = lane & 15;
    const int mbase = hi * 8;
    float* orow = out + (size_t)b * LQ * DD + (size_t)(q0 + mbase) * DD + n;
#pragma unroll
    for (int t = 0; t < 16; ++t) {
#pragma unroll
        for (int j = 0; j < 8; ++j) {
            orow[(size_t)j * DD + t * 16] = acc[t][j];
        }
    }
}

// ---------------------------------------------------------------------------
extern "C" void kernel_launch(void* const* d_in, const int* in_sizes, int n_in,
                              void* d_out, int out_size, void* d_ws, size_t ws_size,
                              hipStream_t stream) {
    const float* x  = (const float*)d_in[0];
    const float* y  = (const float*)d_in[1];
    const unsigned char* mask = (const unsigned char*)d_in[2];   // bool, all-false safe
    const float* Wq = (const float*)d_in[3];
    const float* Wk = (const float*)d_in[4];
    const float* Wv = (const float*)d_in[5];

    float* att_out = (float*)d_out;                              // [B, LQ, D]
    float* attn    = att_out + (size_t)BB * LQ * DD;             // [B, LQ, LK]

    // Workspace carve-up (~48.6 MiB)
    unsigned short* Qbf  = (unsigned short*)d_ws;                // B*LQ*D
    unsigned short* Ktbf = Qbf  + (size_t)BB * LQ * DD;          // B*D*LK
    unsigned short* Vbf  = Ktbf + (size_t)BB * DD * LK;          // B*LK*D
    unsigned short* Wtq  = Vbf  + (size_t)BB * LK * DD;          // D*D each
    unsigned short* Wtk  = Wtq  + (size_t)DD * DD;
    unsigned short* Wtv  = Wtk  + (size_t)DD * DD;
    float* qn = (float*)(Wtv + (size_t)DD * DD);                 // B*LQ
    float* kn = qn + (size_t)BB * LQ;                            // B*LK

    // 1. Weight transpose+convert
    wtrans_kernel<<<(3 * DD * DD) / 256, 256, 0, stream>>>(Wq, Wk, Wv, Wtq, Wtk, Wtv);

    // 2. Projections (Q, Kt, V) via WMMA
    proj_kernel<<<dim3(DD / 16, LQ / 16, BB * 3), 32, 0, stream>>>(
        x, y, Wtq, Wtk, Wtv, Qbf, Ktbf, Vbf);

    // 3-4. Row norms
    qnorm_kernel<<<BB * LQ, 256, 0, stream>>>(Qbf, qn);
    knorm_kernel<<<(BB * LK) / 256, 256, 0, stream>>>(Ktbf, kn);

    // 5. Distance scores -> unnormalized attention probabilities
    scores_kernel<<<dim3(LK / 128, LQ / 16, BB), 32, 0, stream>>>(
        Qbf, Ktbf, qn, kn, mask, attn);

    // 6. Softmax normalize in place
    rownorm_kernel<<<BB * LQ, 256, 0, stream>>>(attn);

    // 7. att_out = attn @ V via WMMA (attn read exactly once)
    av_kernel<<<dim3(LQ / 16, BB), 32, 0, stream>>>(attn, Vbf, att_out);
}
